// Model_52518860096578
// MI455X (gfx1250) — compile-verified, matrix operand order checked by reading, not run on hardware
//
#include <hip/hip_runtime.h>

typedef __attribute__((ext_vector_type(16))) __bf16 v16bf;
typedef __attribute__((ext_vector_type(8)))  float  v8f;
typedef __attribute__((ext_vector_type(4)))  __bf16 v4bf;
typedef __attribute__((ext_vector_type(4)))  unsigned int u32x4;
typedef __attribute__((ext_vector_type(8)))  int i32x8;
typedef __attribute__((ext_vector_type(4)))  int i32x4;

#define NB    4
#define NH    16
#define SDIM  2048
#define DDIM  128
#define QTILE 128   // query rows per workgroup (8 waves x 16)
#define KC    32    // keys per chunk

// ---------------- fp32 -> bf16 conversion of x3 (K == V) ----------------
__global__ void cvt_f32_bf16(const float* __restrict__ x,
                             __bf16* __restrict__ y, long n) {
  long i = ((long)blockIdx.x * blockDim.x + threadIdx.x) * 8;
  if (i + 8 > n) return;
  const float4* p = (const float4*)(x + i);
  float4 a = p[0], b = p[1];
  v4bf lo, hi;
  lo[0] = (__bf16)a.x; lo[1] = (__bf16)a.y; lo[2] = (__bf16)a.z; lo[3] = (__bf16)a.w;
  hi[0] = (__bf16)b.x; hi[1] = (__bf16)b.y; hi[2] = (__bf16)b.z; hi[3] = (__bf16)b.w;
  *(v4bf*)(y + i)     = lo;
  *(v4bf*)(y + i + 4) = hi;
}

// ---------------- DPP butterfly reductions over 16-lane groups ----------------
// ctrls: 0xB1 quad_perm(1,0,3,2)=xor1, 0x4E quad_perm(2,3,0,1)=xor2,
//        0x141 row_half_mirror, 0x140 row_mirror  (all stay inside 16 lanes)
template <int CTRL>
__device__ __forceinline__ float dpp_mov(float x) {
  int t = __builtin_amdgcn_update_dpp(0, __float_as_int(x), CTRL, 0xF, 0xF, true);
  return __int_as_float(t);
}
__device__ __forceinline__ float redmax16(float v) {
  v = fmaxf(v, dpp_mov<0xB1>(v));
  v = fmaxf(v, dpp_mov<0x4E>(v));
  v = fmaxf(v, dpp_mov<0x141>(v));
  v = fmaxf(v, dpp_mov<0x140>(v));
  return v;
}
__device__ __forceinline__ float redsum16(float v) {
  v += dpp_mov<0xB1>(v);
  v += dpp_mov<0x4E>(v);
  v += dpp_mov<0x141>(v);
  v += dpp_mov<0x140>(v);
  return v;
}

// ---------------- TDM: 2-D tensor_load_to_lds (6-arg toolchain variant) ------
__device__ __forceinline__ void tdm_load_2d(unsigned ldsOff, const void* gaddr,
                                            int dataSizeCode, int tileD0,
                                            int tileD1, int tDim0, int tDim1,
                                            long stride0) {
  unsigned long long ga = (unsigned long long)gaddr;
  u32x4 g0;
  g0[0] = 1u;                                   // count=1, user mode
  g0[1] = ldsOff;                               // lds_addr
  g0[2] = (unsigned)(ga & 0xFFFFFFFFu);         // global_addr[31:0]
  g0[3] = (unsigned)((ga >> 32) & 0x1FFFFFFu) | (2u << 30);  // [56:32] | type=2
  i32x8 g1;
  g1[0] = (dataSizeCode << 16);                 // wg_mask=0, data_size
  g1[1] = (int)((tDim0 & 0xFFFF) << 16);        // tensor_dim0[15:0]
  g1[2] = (int)(((tDim0 >> 16) & 0xFFFF) | ((tDim1 & 0xFFFF) << 16));
  g1[3] = (int)(((tDim1 >> 16) & 0xFFFF) | ((tileD0 & 0xFFFF) << 16));
  g1[4] = (int)(tileD1 & 0xFFFF);               // tile_dim1, tile_dim2=0
  g1[5] = (int)(stride0 & 0xFFFFFFFFll);        // tensor_dim0_stride[31:0]
  g1[6] = (int)((stride0 >> 32) & 0xFFFFll);    // stride0[47:32], stride1=0
  g1[7] = 0;
  i32x4 z4; z4[0] = 0; z4[1] = 0; z4[2] = 0; z4[3] = 0;
  i32x8 z8;
  #pragma unroll
  for (int k = 0; k < 8; ++k) z8[k] = 0;
  __builtin_amdgcn_tensor_load_to_lds(g0, g1, z4, z4, z8, 0);
}

// ---------------- flash-attention forward ----------------
template <bool KBF16>
__global__ __launch_bounds__(256)
void fa_fwd(const float* __restrict__ Q, const void* __restrict__ Kv,
            float* __restrict__ O) {
  __shared__ __align__(32) float  qbuf[QTILE * DDIM];     // 64KB, TDM target
  __shared__ __align__(32) __bf16 kvRow[2][KC * DDIM];    // [key][d]  2x8KB
  __shared__ __align__(32) __bf16 kvCol[2][DDIM * KC];    // [d][key]  2x8KB
  __shared__ __align__(32) __bf16 pbuf[8][16 * 32];       // per-wave P, 8KB

  const int tid  = threadIdx.x;
  const int wv   = tid >> 5;
  const int lane = tid & 31;
  const int m16  = lane & 15;
  const int hh   = lane >> 4;

  const int  gid     = blockIdx.x;
  const int  bh      = gid >> 4;          // (b*H + h), 64 total
  const int  qt      = gid & 15;          // 2048/128 q-tiles
  const long headOff = (long)bh * SDIM * DDIM;

  // cooperative K/V loader mapping: 256 threads -> 32 keys x 8 segs of 16 d
  const int lkey = tid >> 3;
  const int lseg = (tid & 7) * 16;
  const int NCH  = SDIM / KC;

  // ---- issue TDM load of the 128x128 f32 Q tile into LDS (wave 0) ----
  if (wv == 0) {
    tdm_load_2d((unsigned)(unsigned long long)(void*)qbuf,
                (const void*)(Q + headOff + (long)qt * QTILE * DDIM),
                /*dataSize 4B*/ 2, DDIM, QTILE, DDIM, QTILE, (long)DDIM);
  }

  // ---- stage + commit K/V chunk 0 while TDM runs ----
  {
    v16bf stg;
    const long src = headOff + (long)lkey * DDIM + lseg;
    if (KBF16) {
      stg = *(const v16bf*)((const __bf16*)Kv + src);
    } else {
      const float* kp = (const float*)Kv + src;
      #pragma unroll
      for (int v = 0; v < 4; ++v) {
        float4 f = ((const float4*)kp)[v];
        stg[v * 4 + 0] = (__bf16)f.x;
        stg[v * 4 + 1] = (__bf16)f.y;
        stg[v * 4 + 2] = (__bf16)f.z;
        stg[v * 4 + 3] = (__bf16)f.w;
      }
    }
    *(v16bf*)&kvRow[0][lkey * DDIM + lseg] = stg;
    #pragma unroll
    for (int j = 0; j < 16; ++j)
      kvCol[0][(lseg + j) * KC + lkey] = stg[j];
  }
  if (wv == 0) __builtin_amdgcn_s_wait_tensorcnt(0);
  __syncthreads();

  // ---- build this wave's Q strip as 4 bf16 A-fragments from LDS ----
  v16bf qf[4];
  {
    const float* qrow = &qbuf[(wv * 16 + m16) * DDIM];
    #pragma unroll
    for (int kb = 0; kb < 4; ++kb) {
      const float4* p = (const float4*)(qrow + kb * 32 + hh * 16);
      #pragma unroll
      for (int v = 0; v < 4; ++v) {
        float4 f = p[v];
        qf[kb][v * 4 + 0] = (__bf16)f.x;
        qf[kb][v * 4 + 1] = (__bf16)f.y;
        qf[kb][v * 4 + 2] = (__bf16)f.z;
        qf[kb][v * 4 + 3] = (__bf16)f.w;
      }
    }
  }

  v8f acc[8];
  #pragma unroll
  for (int d = 0; d < 8; ++d)
    #pragma unroll
    for (int i = 0; i < 8; ++i) acc[d][i] = 0.0f;
  float mrow[8], lrow[8];
  #pragma unroll
  for (int i = 0; i < 8; ++i) { mrow[i] = -3.0e38f; lrow[i] = 0.0f; }

  for (int kc = 0; kc < NCH; ++kc) {
    const int cur = kc & 1;
    const int nxt = cur ^ 1;
    const bool havenext = (kc + 1 < NCH);

    // ---- stage chunk kc+1 into registers (loads issue before compute) ----
    v16bf stg;
    if (havenext) {
      const long src = headOff + (long)((kc + 1) * KC + lkey) * DDIM + lseg;
      if (KBF16) {
        stg = *(const v16bf*)((const __bf16*)Kv + src);
      } else {
        const float* kp = (const float*)Kv + src;
        #pragma unroll
        for (int v = 0; v < 4; ++v) {
          float4 f = ((const float4*)kp)[v];
          stg[v * 4 + 0] = (__bf16)f.x;
          stg[v * 4 + 1] = (__bf16)f.y;
          stg[v * 4 + 2] = (__bf16)f.z;
          stg[v * 4 + 3] = (__bf16)f.w;
        }
      }
    }

    // ---- S = Q * K^T for two 16-wide key tiles ----
    v8f s0, s1;
    #pragma unroll
    for (int i = 0; i < 8; ++i) { s0[i] = 0.0f; s1[i] = 0.0f; }
    #pragma unroll
    for (int kb = 0; kb < 4; ++kb) {
      v16bf b0 = *(const v16bf*)&kvRow[cur][(m16)      * DDIM + kb * 32 + hh * 16];
      v16bf b1 = *(const v16bf*)&kvRow[cur][(16 + m16) * DDIM + kb * 32 + hh * 16];
      s0 = __builtin_amdgcn_wmma_f32_16x16x32_bf16(false, qf[kb], false, b0,
                                                   (short)0, s0, false, false);
      s1 = __builtin_amdgcn_wmma_f32_16x16x32_bf16(false, qf[kb], false, b1,
                                                   (short)0, s1, false, false);
    }

    // ---- online softmax: DPP butterfly reductions (no LDS traffic) ----
    float p0[8], p1[8];
    #pragma unroll
    for (int i = 0; i < 8; ++i) {
      float t  = redmax16(fmaxf(s0[i], s1[i]));
      float mn = fmaxf(mrow[i], t);
      float sc = __expf(mrow[i] - mn);
      mrow[i] = mn;
      p0[i] = __expf(s0[i] - mn);
      p1[i] = __expf(s1[i] - mn);
      float r = redsum16(p0[i] + p1[i]);
      lrow[i] = lrow[i] * sc + r;
      #pragma unroll
      for (int d = 0; d < 8; ++d) acc[d][i] *= sc;
    }

    // ---- transpose P (C-layout) -> bf16 A-layout via per-wave LDS ----
    {
      __bf16* pb = pbuf[wv];
      #pragma unroll
      for (int i = 0; i < 8; ++i) {
        int row = i + hh * 8;
        pb[row * 32 + m16]      = (__bf16)p0[i];
        pb[row * 32 + 16 + m16] = (__bf16)p1[i];
      }
    }
    v16bf pf = *(const v16bf*)&pbuf[wv][m16 * 32 + hh * 16];

    // ---- O += P * V : 8 D-tiles, K=32 ----
    #pragma unroll
    for (int dt = 0; dt < 8; ++dt) {
      v16bf vf = *(const v16bf*)&kvCol[cur][(dt * 16 + m16) * KC + hh * 16];
      acc[dt] = __builtin_amdgcn_wmma_f32_16x16x32_bf16(false, pf, false, vf,
                                                        (short)0, acc[dt],
                                                        false, false);
    }

    __syncthreads();   // prior readers of buf[nxt] are done
    if (havenext) {
      *(v16bf*)&kvRow[nxt][lkey * DDIM + lseg] = stg;
      #pragma unroll
      for (int j = 0; j < 16; ++j)
        kvCol[nxt][(lseg + j) * KC + lkey] = stg[j];
    }
    __syncthreads();   // buf[nxt] ready for next iteration
  }

  // ---- normalize and store ----
  const int qr0 = qt * QTILE + wv * 16;
  float* orow = O + headOff + (long)qr0 * DDIM;
  #pragma unroll
  for (int i = 0; i < 8; ++i) {
    float inv = 1.0f / lrow[i];
    int row = i + hh * 8;
    #pragma unroll
    for (int dt = 0; dt < 8; ++dt)
      orow[(long)row * DDIM + dt * 16 + m16] = acc[dt][i] * inv;
  }
}

extern "C" void kernel_launch(void* const* d_in, const int* in_sizes, int n_in,
                              void* d_out, int out_size, void* d_ws, size_t ws_size,
                              hipStream_t stream) {
  const float* x2 = (const float*)d_in[0];   // Q
  const float* x3 = (const float*)d_in[1];   // K == V
  float* out = (float*)d_out;

  const long n = (long)NB * NH * SDIM * DDIM;        // 16,777,216
  const int  grid = NB * NH * (SDIM / QTILE);        // 1024 workgroups

  if (ws_size >= (size_t)n * sizeof(__bf16)) {
    __bf16* kbf = (__bf16*)d_ws;
    long thr = n / 8;
    cvt_f32_bf16<<<(int)((thr + 255) / 256), 256, 0, stream>>>(x3, kbf, n);
    fa_fwd<true><<<grid, 256, 0, stream>>>(x2, (const void*)kbf, out);
  } else {
    fa_fwd<false><<<grid, 256, 0, stream>>>(x2, (const void*)x3, out);
  }
}
